// SimilarLearner_Aggregator_12120397709896
// MI455X (gfx1250) — matrix operand order repaired and checked by reading, not compiled, not run on hardware
//
#include <hip/hip_runtime.h>

typedef _Float16 half_t;
typedef __attribute__((ext_vector_type(16))) _Float16 v16h;
typedef __attribute__((ext_vector_type(8)))  float    v8f;
typedef __attribute__((ext_vector_type(4)))  _Float16 h4;

#define DEG       50
#define EMBED     64
#define BATCH_PB  8                    // batch elements per block
#define ROWS      (BATCH_PB * DEG)     // 400 MLP rows per block
#define NTILES    (ROWS / 16)          // 25 M-tiles of 16 rows
#define XS        72                   // padded LDS stride (halves) for ens/hs
#define W1S       136                  // padded stride for transposed w1 (K=128)
#define W2S       72                   // padded stride for transposed w2 (K=64)

__global__ __launch_bounds__(256, 1)
void graphrec_agg_kernel(const int* __restrict__ nodes,
                         const int* __restrict__ neighbors,
                         const float* __restrict__ u_to_e,
                         const float* __restrict__ w1, const float* __restrict__ b1,
                         const float* __restrict__ w2, const float* __restrict__ b2,
                         const float* __restrict__ w3, const float* __restrict__ b3,
                         float* __restrict__ out) {
    __shared__ __align__(16) half_t ens[ROWS * XS];       // gathered neighbor embeds (f16)
    __shared__ __align__(16) half_t hs [ROWS * XS];       // hidden activations (f16)
    __shared__ __align__(16) half_t eus[BATCH_PB * EMBED];// gathered self embeds (f16)
    __shared__ __align__(16) half_t w1t[EMBED * W1S];     // w1 transposed [N][K=128]
    __shared__ __align__(16) half_t w2t[EMBED * W2S];     // w2 transposed [N][K=64]
    __shared__ float b1s[EMBED], b2s[EMBED], w3s[EMBED];
    __shared__ float sc[ROWS];
    __shared__ float b3s;

    const int tid  = threadIdx.x;
    const int wave = tid >> 5;
    const int lane = tid & 31;
    const int g    = lane >> 4;    // lane half-group (0/1)
    const int nloc = lane & 15;    // N column / A row within tile
    const long b0  = (long)blockIdx.x * BATCH_PB;

    // ---- Stage 0: gather embeddings + stage weights into LDS (f16) ----
    // neighbor rows: neighbors is [B,50] row-major -> block rows are contiguous
    for (int i = tid; i < ROWS * 16; i += 256) {          // 16 float4 per 64-col row
        int r = i >> 4, q = i & 15;
        long nbr = neighbors[b0 * DEG + r];
        float4 v = ((const float4*)(u_to_e + nbr * EMBED))[q];
        h4 hv = { (half_t)v.x, (half_t)v.y, (half_t)v.z, (half_t)v.w };
        *(h4*)&ens[r * XS + q * 4] = hv;
    }
    for (int i = tid; i < BATCH_PB * 16; i += 256) {
        int b = i >> 4, q = i & 15;
        long nd = nodes[b0 + b];
        float4 v = ((const float4*)(u_to_e + nd * EMBED))[q];
        h4 hv = { (half_t)v.x, (half_t)v.y, (half_t)v.z, (half_t)v.w };
        *(h4*)&eus[b * EMBED + q * 4] = hv;
    }
    for (int i = tid; i < 128 * 64; i += 256) {           // w1 [128,64] -> w1t[n][k]
        int k = i >> 6, n = i & 63;
        w1t[n * W1S + k] = (half_t)w1[k * 64 + n];
    }
    for (int i = tid; i < 64 * 64; i += 256) {            // w2 [64,64] -> w2t[n][k]
        int k = i >> 6, n = i & 63;
        w2t[n * W2S + k] = (half_t)w2[k * 64 + n];
    }
    if (tid < 64) { b1s[tid] = b1[tid]; b2s[tid] = b2[tid]; w3s[tid] = w3[tid]; }
    if (tid == 0) b3s = b3[0];
    __syncthreads();

    // ---- Stage 1+2: per-wave MLP over 16-row tiles via WMMA ----
    for (int t = wave; t < NTILES; t += 8) {
        const int arow = t * 16 + nloc;        // this lane's A-matrix row
        const int rb   = arow / DEG;           // batch-local index of that row

        // ---- layer 1: [16x128] x [128x64], K in 4 steps of 32 ----
        v8f acc[4] = {v8f{}, v8f{}, v8f{}, v8f{}};
        #pragma unroll
        for (int kk = 0; kk < 4; ++kk) {
            v16h a;
            uint4* au = (uint4*)&a;
            // A layout: halves 0..7 -> K = g*8+0..7 ; halves 8..15 -> K = 16+g*8+0..7
            if (kk < 2) {
                const half_t* src = &ens[arow * XS + kk * 32 + g * 8];
                au[0] = *(const uint4*)(src);
                au[1] = *(const uint4*)(src + 16);
            } else {
                const half_t* src = &eus[rb * EMBED + (kk - 2) * 32 + g * 8];
                au[0] = *(const uint4*)(src);
                au[1] = *(const uint4*)(src + 16);
            }
            const int kb = kk * 32 + g * 8;    // B layout: halves 0..7 -> K = g*8+0..7 ; halves 8..15 -> K = 16+g*8+0..7
            #pragma unroll
            for (int j = 0; j < 4; ++j) {
                v16h bm;
                uint4* bu = (uint4*)&bm;
                const half_t* wr = &w1t[(j * 16 + nloc) * W1S + kb];
                bu[0] = *(const uint4*)(wr);
                bu[1] = *(const uint4*)(wr + 16);
                acc[j] = __builtin_amdgcn_wmma_f32_16x16x32_f16(
                    false, a, false, bm, (short)0, acc[j], false, false);
            }
        }
        // epilogue: bias + relu -> f16 hidden in LDS
        #pragma unroll
        for (int j = 0; j < 4; ++j) {
            int n = j * 16 + nloc;
            float bias = b1s[n];
            #pragma unroll
            for (int r = 0; r < 8; ++r) {
                float v = acc[j][r] + bias;
                v = v > 0.f ? v : 0.f;
                hs[(t * 16 + r + g * 8) * XS + n] = (half_t)v;
            }
        }

        // ---- layer 2: [16x64] x [64x64], K in 2 steps of 32 (in-place relu) ----
        v8f acc2[4] = {v8f{}, v8f{}, v8f{}, v8f{}};
        #pragma unroll
        for (int kk = 0; kk < 2; ++kk) {
            v16h a;
            uint4* au = (uint4*)&a;
            const half_t* src = &hs[arow * XS + kk * 32 + g * 8];
            au[0] = *(const uint4*)(src);
            au[1] = *(const uint4*)(src + 16);
            const int kb = kk * 32 + g * 8;
            #pragma unroll
            for (int j = 0; j < 4; ++j) {
                v16h bm;
                uint4* bu = (uint4*)&bm;
                const half_t* wr = &w2t[(j * 16 + nloc) * W2S + kb];
                bu[0] = *(const uint4*)(wr);
                bu[1] = *(const uint4*)(wr + 16);
                acc2[j] = __builtin_amdgcn_wmma_f32_16x16x32_f16(
                    false, a, false, bm, (short)0, acc2[j], false, false);
            }
        }
        #pragma unroll
        for (int j = 0; j < 4; ++j) {
            int n = j * 16 + nloc;
            float bias = b2s[n];
            #pragma unroll
            for (int r = 0; r < 8; ++r) {
                float v = acc2[j][r] + bias;
                v = v > 0.f ? v : 0.f;
                hs[(t * 16 + r + g * 8) * XS + n] = (half_t)v;
            }
        }
    }
    __syncthreads();

    // ---- Stage 3: attention scores = h2 . w3 + b3 ----
    for (int r = tid; r < ROWS; r += 256) {
        float s = b3s;
        #pragma unroll 8
        for (int n = 0; n < EMBED; ++n) s += (float)hs[r * XS + n] * w3s[n];
        sc[r] = s;
    }
    __syncthreads();

    // ---- Stage 4: softmax over DEG=50 neighbors (wave per batch element) ----
    {
        const int base = wave * DEG;           // 8 waves <-> 8 batch elements
        float s1 = sc[base + lane];
        bool  v2 = (lane + 32) < DEG;
        float s2 = v2 ? sc[base + lane + 32] : -__builtin_inff();
        float mx = fmaxf(s1, s2);
        #pragma unroll
        for (int off = 16; off > 0; off >>= 1) mx = fmaxf(mx, __shfl_xor(mx, off, 32));
        float e1 = __expf(s1 - mx);
        float e2 = v2 ? __expf(s2 - mx) : 0.f;
        float sm = e1 + e2;
        #pragma unroll
        for (int off = 16; off > 0; off >>= 1) sm += __shfl_xor(sm, off, 32);
        float inv = 1.f / sm;
        sc[base + lane] = e1 * inv;
        if (v2) sc[base + lane + 32] = e2 * inv;
    }
    __syncthreads();

    // ---- Stage 5: out[b,:] = sum_d att[b,d] * e_n[b,d,:] ----
    for (int i = tid; i < BATCH_PB * EMBED; i += 256) {
        int b = i >> 6, e = i & 63;
        const half_t* er = &ens[b * DEG * XS + e];
        const float*  ar = &sc[b * DEG];
        float acc = 0.f;
        #pragma unroll 5
        for (int d = 0; d < DEG; ++d) acc += ar[d] * (float)er[d * XS];
        out[(b0 + b) * EMBED + e] = acc;
    }
}

extern "C" void kernel_launch(void* const* d_in, const int* in_sizes, int n_in,
                              void* d_out, int out_size, void* d_ws, size_t ws_size,
                              hipStream_t stream) {
    const int*   nodes     = (const int*)d_in[0];
    const int*   neighbors = (const int*)d_in[1];
    const float* u_to_e    = (const float*)d_in[2];
    const float* w1        = (const float*)d_in[3];
    const float* b1        = (const float*)d_in[4];
    const float* w2        = (const float*)d_in[5];
    const float* b2        = (const float*)d_in[6];
    const float* w3        = (const float*)d_in[7];
    const float* b3        = (const float*)d_in[8];
    float*       out       = (float*)d_out;

    const int batch  = in_sizes[0];            // 16384
    const int blocks = batch / BATCH_PB;       // 2048
    graphrec_agg_kernel<<<blocks, 256, 0, stream>>>(
        nodes, neighbors, u_to_e, w1, b1, w2, b2, w3, b3, out);
}
